// OptimizedAttention_54219667145208
// MI455X (gfx1250) — matrix operand order repaired and checked by reading, not compiled, not run on hardware
//
#include <hip/hip_runtime.h>
#include <hip/hip_bf16.h>

typedef unsigned short u16;
typedef unsigned int   u32;
typedef __attribute__((ext_vector_type(16))) __bf16 v16bf;
typedef __attribute__((ext_vector_type(8)))  float  v8f;
typedef int vint4 __attribute__((vector_size(16)));   // matches async-builtin pointee

#define B_SZ  2
#define T_SZ  2048
#define C_SZ  2048
#define NH    16
#define NKV   4
#define HD    128
#define QKV_N 3072   /* (16 + 2*4) * 128 */
#define GROUP 4      /* NH / NKV */

// ---- CDNA5 async VMEM->LDS path (guarded; falls back to sync copy) ----
#if defined(__has_builtin)
# if __has_builtin(__builtin_amdgcn_global_load_async_to_lds_b128) && \
     __has_builtin(__builtin_amdgcn_s_wait_asynccnt)
#  define ASYNC_OK 1
# endif
#endif
#ifndef ASYNC_OK
# define ASYNC_OK 0
#endif

// copy 16 bytes (8 bf16) global -> LDS
__device__ __forceinline__ void copy16_to_lds(u16* lds, const u16* g) {
#if ASYNC_OK
  __builtin_amdgcn_global_load_async_to_lds_b128(
      (__attribute__((address_space(1))) vint4*)g,
      (__attribute__((address_space(3))) vint4*)lds, 0, 0);
#else
  *(uint4*)lds = *(const uint4*)g;
#endif
}

__device__ __forceinline__ void async_fence() {
#if ASYNC_OK
  __builtin_amdgcn_s_wait_asynccnt(0);
#endif
}

// ---------- helpers ----------

__device__ __forceinline__ u16 f2bf(float f) {
  u32 x = __float_as_uint(f);
  x += 0x7FFFu + ((x >> 16) & 1u);   // round-to-nearest-even
  return (u16)(x >> 16);
}

// WMMA A-fragment (16x32 bf16, M x K) from row-major [16][stride] u16 storage.
// ISA layout: lane half 0 -> K 0..7 and 16..23 ; half 1 -> K 8..15 and 24..31.
__device__ __forceinline__ v16bf frag_a(const u16* p, int stride) {
  int lane = threadIdx.x & 31;
  int row  = lane & 15;
  int kb   = (lane >> 4) << 3;        // 0 or 8
  const u16* r0 = p + row * stride + kb;
  union { uint4 u[2]; v16bf v; } t;
  t.u[0] = *(const uint4*)(r0);
  t.u[1] = *(const uint4*)(r0 + 16);
  return t.v;
}

// WMMA B-fragment (32x16 bf16, K x N) from N-major [16][stride] u16 storage
// (each row = one output column, elements along K).
// ISA layout: lane half 0 -> K 0..7 and 16..23 ; half 1 -> K 8..15 and 24..31.
__device__ __forceinline__ v16bf frag_b(const u16* p, int stride) {
  int lane = threadIdx.x & 31;
  int col  = lane & 15;
  int kb   = (lane >> 4) << 3;        // 0 or 8
  const u16* r0 = p + col * stride + kb;
  union { uint4 u[2]; v16bf v; } t;
  t.u[0] = *(const uint4*)(r0);
  t.u[1] = *(const uint4*)(r0 + 16);
  return t.v;
}

__device__ __forceinline__ v8f wmma_bf16(v16bf a, v16bf b, v8f c) {
  return __builtin_amdgcn_wmma_f32_16x16x32_bf16(false, a, false, b,
                                                 (short)0, c, false, false);
}

// ---------- kernels ----------

__global__ void k_f32_to_bf16(const float* __restrict__ src,
                              u16* __restrict__ dst, int n) {
  int i = blockIdx.x * blockDim.x + threadIdx.x;
  if (i < n) dst[i] = f2bf(src[i]);
}

// C[M,N] (f32) = A[M,K] (bf16, row-major) * B[K,N] (bf16, row-major)
// M%128==0, N%128==0, K%32==0. 256 threads = 8 waves; block tile 128x128.
// Wave (mw,nw) with mw=w>>1, nw=w&1 computes rows [mw*32,+32) x cols [nw*64,+64):
// 2 A-frags x 4 B-frags -> 8 WMMAs per 32-deep K chunk.
__global__ __launch_bounds__(256)
void k_gemm_bf16(const u16* __restrict__ A, const u16* __restrict__ Bm,
                 float* __restrict__ C, int M, int N, int K) {
  __shared__ u16 As[128 * 32];  // [128 rows][32 k]
  __shared__ u16 Bt[128 * 32];  // [128 cols][32 k]  (transposed on fill)
  int tid  = threadIdx.x;
  int wave = tid >> 5, lane = tid & 31;
  int mw = wave >> 1, nw = wave & 1;
  int ntiles = N / 128;
  int m0 = (blockIdx.x / ntiles) * 128;
  int n0 = (blockIdx.x % ntiles) * 128;

  v8f z = {};
  v8f acc[2][4] = { { z, z, z, z }, { z, z, z, z } };

  for (int k0 = 0; k0 < K; k0 += 32) {
    {   // A tile (async VMEM->LDS): thread -> row tid/2, 16-elem segment tid%2
      int row = tid >> 1, seg = tid & 1;
      const u16* g = A + (size_t)(m0 + row) * K + k0 + seg * 16;
      u16* s = As + row * 32 + seg * 16;
      copy16_to_lds(s, g);
      copy16_to_lds(s + 8, g + 8);
    }
    {   // B tile, transposed: thread -> k-row tid%32, 16-col segment tid/32
      int k = tid & 31, seg = tid >> 5;
      const u16* g = Bm + (size_t)(k0 + k) * N + n0 + seg * 16;
      alignas(16) u16 tmp[16];
      *(uint4*)(tmp)     = *(const uint4*)(g);
      *(uint4*)(tmp + 8) = *(const uint4*)(g + 8);
#pragma unroll
      for (int j = 0; j < 16; ++j)
        Bt[(seg * 16 + j) * 32 + k] = tmp[j];
    }
    async_fence();
    __syncthreads();

    // preload all fragments, then a dense WMMA chain
    v16bf af[2], bf[4];
#pragma unroll
    for (int i = 0; i < 2; ++i)
      af[i] = frag_a(As + (mw * 32 + i * 16) * 32, 32);
#pragma unroll
    for (int nt = 0; nt < 4; ++nt)
      bf[nt] = frag_b(Bt + (nw * 64 + nt * 16) * 32, 32);
#pragma unroll
    for (int i = 0; i < 2; ++i)
#pragma unroll
      for (int nt = 0; nt < 4; ++nt)
        acc[i][nt] = wmma_bf16(af[i], bf[nt], acc[i][nt]);
    __syncthreads();
  }

  // C/D layout: lane holds N = lane%16 ; VGPR r holds M = r + 8*(lane/16)
  int hh = lane >> 4, col = lane & 15;
#pragma unroll
  for (int i = 0; i < 2; ++i)
#pragma unroll
    for (int nt = 0; nt < 4; ++nt)
#pragma unroll
      for (int r = 0; r < 8; ++r) {
        int m = m0 + mw * 32 + i * 16 + r + 8 * hh;
        C[(size_t)m * N + n0 + nw * 64 + nt * 16 + col] = acc[i][nt][r];
      }
}

// Split QKV (f32) -> head-major bf16 Q/K/V with interleaved-pair RoPE on Q,K.
__global__ void k_rope_split(const float* __restrict__ qkv,
                             const float* __restrict__ cache,
                             u16* __restrict__ qb, u16* __restrict__ kb,
                             u16* __restrict__ vb) {
  int gid  = blockIdx.x * blockDim.x + threadIdx.x;
  int pair = gid & 63;
  int rest = gid >> 6;
  int head = rest % 24;
  rest /= 24;
  int t = rest % T_SZ;
  int b = rest / T_SZ;
  if (b >= B_SZ) return;

  const float* row = qkv + (size_t)(b * T_SZ + t) * QKV_N;
  float c = cache[(t * 64 + pair) * 2 + 0];
  float s = cache[(t * 64 + pair) * 2 + 1];

  if (head < NH) {
    float x0 = row[head * HD + pair * 2];
    float x1 = row[head * HD + pair * 2 + 1];
    size_t o = (((size_t)(b * NH + head)) * T_SZ + t) * HD + pair * 2;
    qb[o]     = f2bf(x0 * c - x1 * s);
    qb[o + 1] = f2bf(x0 * s + x1 * c);
  } else if (head < NH + NKV) {
    int h = head - NH;
    float x0 = row[NH * HD + h * HD + pair * 2];
    float x1 = row[NH * HD + h * HD + pair * 2 + 1];
    size_t o = (((size_t)(b * NKV + h)) * T_SZ + t) * HD + pair * 2;
    kb[o]     = f2bf(x0 * c - x1 * s);
    kb[o + 1] = f2bf(x0 * s + x1 * c);
  } else {
    int h = head - NH - NKV;
    float x0 = row[(NH + NKV) * HD + h * HD + pair * 2];
    float x1 = row[(NH + NKV) * HD + h * HD + pair * 2 + 1];
    size_t o = (((size_t)(b * NKV + h)) * T_SZ + t) * HD + pair * 2;
    vb[o]     = f2bf(x0);
    vb[o + 1] = f2bf(x1);
  }
}

// Flash attention, causal, GQA. Block = one (b, h, 64-row q tile); 4 waves,
// wave owns 16 q rows. KV tiles of 32 keys through LDS; online softmax.
__global__ __launch_bounds__(128)
void k_flash_attn(const u16* __restrict__ qb, const u16* __restrict__ kb,
                  const u16* __restrict__ vb, u16* __restrict__ ob) {
  __shared__ u16 Ks[32 * 128];      // [key][d]
  __shared__ u16 Vt[128 * 32];      // [d][key]  (transposed on fill)
  __shared__ u16 Ps[4 * 16 * 32];   // per-wave P staging [16 q][32 key]

  int tid = threadIdx.x, wave = tid >> 5, lane = tid & 31;
  int qtiles = T_SZ / 64;
  int qt = blockIdx.x % qtiles;
  int h  = (blockIdx.x / qtiles) % NH;
  int b  =  blockIdx.x / qtiles / NH;
  int hk = h / GROUP;
  int q0 = qt * 64 + wave * 16;

  const u16* Qb = qb + ((size_t)(b * NH  + h )) * T_SZ * HD;
  const u16* Kb = kb + ((size_t)(b * NKV + hk)) * T_SZ * HD;
  const u16* Vb = vb + ((size_t)(b * NKV + hk)) * T_SZ * HD;

  // Q fragments (global layout already matches A-frag addressing)
  v16bf qf[4];
  {
    int row = lane & 15, kbo = (lane >> 4) << 3;
    const u16* qp = Qb + (size_t)(q0 + row) * HD;
#pragma unroll
    for (int c2 = 0; c2 < 4; ++c2) {
      union { uint4 u[2]; v16bf v; } t;
      t.u[0] = *(const uint4*)(qp + c2 * 32 + kbo);
      t.u[1] = *(const uint4*)(qp + c2 * 32 + kbo + 16);
      qf[c2] = t.v;
    }
  }

  v8f z = {};
  v8f o[8] = { z, z, z, z, z, z, z, z };
  float mrow[8], lrow[8];
#pragma unroll
  for (int r = 0; r < 8; ++r) { mrow[r] = -1e30f; lrow[r] = 0.f; }

  int hh = lane >> 4, col = lane & 15;
  const float scale = 0.08838834764831845f;   // 1/sqrt(128)
  u16* Pw = Ps + wave * 16 * 32;
  int kend = qt * 64 + 64;                    // causal: keys < kend

  for (int kv0 = 0; kv0 < kend; kv0 += 32) {
    {   // K tile fill (async VMEM->LDS): thread -> key tid/4, 32-elem segment tid%4
      int row = tid >> 2, seg = tid & 3;
      const u16* g = Kb + (size_t)(kv0 + row) * HD + seg * 32;
      u16* s = Ks + row * 128 + seg * 32;
      copy16_to_lds(s,      g);
      copy16_to_lds(s + 8,  g + 8);
      copy16_to_lds(s + 16, g + 16);
      copy16_to_lds(s + 24, g + 24);
    }
    {   // V tile fill (transpose): thread -> key tid%32, d segment tid/32
      int key = tid & 31, dseg = tid >> 5;
      const u16* g = Vb + (size_t)(kv0 + key) * HD + dseg * 32;
      alignas(16) u16 tmp[32];
      *(uint4*)(tmp)      = *(const uint4*)(g);
      *(uint4*)(tmp + 8)  = *(const uint4*)(g + 8);
      *(uint4*)(tmp + 16) = *(const uint4*)(g + 16);
      *(uint4*)(tmp + 24) = *(const uint4*)(g + 24);
#pragma unroll
      for (int j = 0; j < 32; ++j)
        Vt[(dseg * 32 + j) * 32 + key] = tmp[j];
    }
    async_fence();
    __syncthreads();

    // S = Q * K^T  (16 x 32), fp32 accum
    v8f s0 = z, s1 = z;
#pragma unroll
    for (int c2 = 0; c2 < 4; ++c2) {
      v16bf k0f = frag_b(Ks +  0 * 128 + c2 * 32, 128);  // keys 0..15
      v16bf k1f = frag_b(Ks + 16 * 128 + c2 * 32, 128);  // keys 16..31
      s0 = wmma_bf16(qf[c2], k0f, s0);
      s1 = wmma_bf16(qf[c2], k1f, s1);
    }

    // online softmax (each wave owns its rows; 16-lane reductions)
    float pfac[8];
#pragma unroll
    for (int r = 0; r < 8; ++r) {
      int qg = q0 + r + 8 * hh;
      float v0 = s0[r] * scale, v1 = s1[r] * scale;
      if (kv0 + col      > qg) v0 = -3.0e38f;
      if (kv0 + 16 + col > qg) v1 = -3.0e38f;
      float mx = fmaxf(v0, v1);
#pragma unroll
      for (int d2 = 1; d2 < 16; d2 <<= 1) mx = fmaxf(mx, __shfl_xor(mx, d2, 32));
      float mnew = fmaxf(mrow[r], mx);
      float p0 = __expf(v0 - mnew);
      float p1 = __expf(v1 - mnew);
      float alpha = __expf(mrow[r] - mnew);
      float ps = p0 + p1;
#pragma unroll
      for (int d2 = 1; d2 < 16; d2 <<= 1) ps += __shfl_xor(ps, d2, 32);
      lrow[r] = lrow[r] * alpha + ps;
      mrow[r] = mnew;
      pfac[r] = alpha;
      s0[r] = p0; s1[r] = p1;
    }
#pragma unroll
    for (int dt = 0; dt < 8; ++dt)
#pragma unroll
      for (int r = 0; r < 8; ++r) o[dt][r] *= pfac[r];

    // stage P (C-layout regs -> A-layout LDS), bf16
#pragma unroll
    for (int r = 0; r < 8; ++r) {
      int mloc = r + 8 * hh;
      Pw[mloc * 32 + col]      = f2bf(s0[r]);
      Pw[mloc * 32 + 16 + col] = f2bf(s1[r]);
    }

    // O += P * V
    v16bf pf = frag_a(Pw, 32);
#pragma unroll
    for (int dt = 0; dt < 8; ++dt) {
      v16bf vf = frag_b(Vt + dt * 16 * 32, 32);
      o[dt] = wmma_bf16(pf, vf, o[dt]);
    }
    __syncthreads();
  }

  // normalize + write attention output as bf16 [B*T, NH*HD]
#pragma unroll
  for (int dt = 0; dt < 8; ++dt)
#pragma unroll
    for (int r = 0; r < 8; ++r) {
      int qrow = q0 + r + 8 * hh;
      float val = o[dt][r] / lrow[r];
      ob[((size_t)(b * T_SZ + qrow)) * C_SZ + h * HD + dt * 16 + col] = f2bf(val);
    }
}

// ---------- launch ----------

extern "C" void kernel_launch(void* const* d_in, const int* in_sizes, int n_in,
                              void* d_out, int out_size, void* d_ws, size_t ws_size,
                              hipStream_t stream) {
  const float* x     = (const float*)d_in[0];
  const float* rope  = (const float*)d_in[1];
  const float* w_qkv = (const float*)d_in[2];
  const float* w_o   = (const float*)d_in[3];
  float* out = (float*)d_out;

  char* ws = (char*)d_ws;
  size_t off = 0;
  auto alloc = [&](size_t bytes) {
    void* p = ws + off;
    off += (bytes + 255) & ~(size_t)255;
    return p;
  };
  u16*   xb   = (u16*)  alloc((size_t)B_SZ * T_SZ * C_SZ  * 2);
  u16*   wqb  = (u16*)  alloc((size_t)C_SZ * QKV_N        * 2);
  u16*   wob  = (u16*)  alloc((size_t)C_SZ * C_SZ         * 2);
  float* qkv  = (float*)alloc((size_t)B_SZ * T_SZ * QKV_N * 4);
  u16*   qb   = (u16*)  alloc((size_t)B_SZ * NH  * T_SZ * HD * 2);
  u16*   kbp  = (u16*)  alloc((size_t)B_SZ * NKV * T_SZ * HD * 2);
  u16*   vbp  = (u16*)  alloc((size_t)B_SZ * NKV * T_SZ * HD * 2);
  u16*   attb = (u16*)  alloc((size_t)B_SZ * T_SZ * C_SZ  * 2);

  int n1 = B_SZ * T_SZ * C_SZ;
  k_f32_to_bf16<<<(n1 + 255) / 256, 256, 0, stream>>>(x, xb, n1);
  int n2 = C_SZ * QKV_N;
  k_f32_to_bf16<<<(n2 + 255) / 256, 256, 0, stream>>>(w_qkv, wqb, n2);
  int n3 = C_SZ * C_SZ;
  k_f32_to_bf16<<<(n3 + 255) / 256, 256, 0, stream>>>(w_o, wob, n3);

  {   // QKV projection: M=4096, N=3072, K=2048
    int M = B_SZ * T_SZ, N = QKV_N, K = C_SZ;
    k_gemm_bf16<<<dim3((M / 128) * (N / 128)), 256, 0, stream>>>(xb, wqb, qkv, M, N, K);
  }
  {   // RoPE + split
    int total = B_SZ * T_SZ * 24 * 64;
    k_rope_split<<<(total + 255) / 256, 256, 0, stream>>>(qkv, rope, qb, kbp, vbp);
  }
  {   // attention
    k_flash_attn<<<dim3(B_SZ * NH * (T_SZ / 64)), 128, 0, stream>>>(qb, kbp, vbp, attb);
  }
  {   // output projection: M=4096, N=2048, K=2048
    int M = B_SZ * T_SZ, N = C_SZ, K = C_SZ;
    k_gemm_bf16<<<dim3((M / 128) * (N / 128)), 256, 0, stream>>>(attb, wob, out, M, N, K);
  }
}
